// EuclideanCodebook_47811575939675
// MI455X (gfx1250) — matrix operand order repaired and checked by reading, not compiled, not run on hardware
//
#include <hip/hip_runtime.h>

typedef __attribute__((ext_vector_type(16))) __bf16        v16bf;
typedef __attribute__((ext_vector_type(8)))  float         v8f;
typedef __attribute__((ext_vector_type(8)))  unsigned int  v8u;
typedef __attribute__((ext_vector_type(4)))  float         v4f;

#define D_DIM            64
#define WAVES_PER_BLOCK  4
#define ROWS_PER_WAVE    32                       // two 16-row WMMA sets
#define CODES_PER_CHUNK  64
#define TILES_PER_CHUNK  (CODES_PER_CHUNK / 16)   // 4
#define CHUNK_BYTES      (CODES_PER_CHUNK * 256)  // 16 KB (hi|lo records)
#define E2_BYTES         (CODES_PER_CHUNK * 4)    // 256 B
#define BUF_BYTES        (CHUNK_BYTES + E2_BYTES)

// ---- fp32 <-> bf16 (round-to-nearest-even) via bit ops ----
__device__ __forceinline__ unsigned f32_to_bf16_rne(float f) {
  unsigned u = __float_as_uint(f);
  unsigned r = 0x7FFFu + ((u >> 16) & 1u);
  return (u + r) >> 16;
}
__device__ __forceinline__ float bf16_bits_to_f32(unsigned b) {
  return __uint_as_float(b << 16);
}

// ---- CDNA5 async global->LDS copy (16 B per lane), ASYNCcnt-tracked ----
__device__ __forceinline__ void async_copy_b128(const void* gsrc, void* ldst) {
  unsigned           lofs = (unsigned)(size_t)ldst;       // LDS byte offset
  unsigned long long ga   = (unsigned long long)(size_t)gsrc;
  asm volatile("global_load_async_to_lds_b128 %0, %1, off"
               :: "v"(lofs), "v"(ga)
               : "memory");
}
__device__ __forceinline__ void wait_asynccnt0() {
  asm volatile("s_wait_asynccnt 0x0" ::: "memory");
}

// ----------------------------------------------------------------------------
// Prep: interleaved bf16 record per code: [hi x64 | lo x64] (256 B), plus e2.
// ----------------------------------------------------------------------------
__global__ void prep_codebook(const float* __restrict__ embed,
                              unsigned short* __restrict__ ecomb,
                              float* __restrict__ e2) {
  const int k = blockIdx.x;
  const int t = threadIdx.x;                 // 0..63
  const float v = embed[(size_t)k * D_DIM + t];
  const unsigned hb = f32_to_bf16_rne(v);
  const float    hf = bf16_bits_to_f32(hb);
  const unsigned lb = f32_to_bf16_rne(v - hf);
  ecomb[(size_t)k * 128 + t]      = (unsigned short)hb;
  ecomb[(size_t)k * 128 + 64 + t] = (unsigned short)lb;

  float s = v * v;
  #pragma unroll
  for (int m = 16; m >= 1; m >>= 1) s += __shfl_xor(s, m, 32);
  __shared__ float partial[2];
  if ((t & 31) == 0) partial[t >> 5] = s;
  __syncthreads();
  if (t == 0) e2[k] = partial[0] + partial[1];
}

// Issue async copies for one codebook chunk (+ its e2 slice) into LDS buffer.
__device__ __forceinline__ void issue_chunk_copy(const unsigned char* __restrict__ gsrc,
                                                 const float* __restrict__ e2src,
                                                 unsigned char* lbuf,
                                                 int wave, int lane) {
  const int quarter = CHUNK_BYTES / WAVES_PER_BLOCK;        // 4096 B per wave
  const unsigned char* gp = gsrc + wave * quarter + lane * 16;
  unsigned char*       lp = lbuf + wave * quarter + lane * 16;
  #pragma unroll
  for (int i = 0; i < quarter / 512; ++i)                   // 8 per wave
    async_copy_b128(gp + i * 512, lp + i * 512);
  if (wave == 0 && lane < (E2_BYTES / 16))                  // 16 lanes x 16 B
    async_copy_b128((const unsigned char*)e2src + lane * 16,
                    lbuf + CHUNK_BYTES + lane * 16);
}

// ----------------------------------------------------------------------------
// Main: each wave owns 32 rows of x (two 16-row WMMA sets). Codebook streams
// through a double-buffered LDS chunk pipeline (async copies, ASYNCcnt).
// score[k] = ||e_k||^2 - 2*x.e_k ; x.e via bf16 hi/lo split (3 WMMAs / chunk).
// ----------------------------------------------------------------------------
__global__ __launch_bounds__(WAVES_PER_BLOCK * 32)
void vq_assign(const float* __restrict__ x,
               const float* __restrict__ embed,
               const unsigned char* __restrict__ ecombB,
               const float* __restrict__ e2,
               float* __restrict__ out,
               int numChunks) {
  __shared__ __align__(16) unsigned char sbuf[2][BUF_BYTES];

  const int tid  = threadIdx.x;
  const int lane = tid & 31;
  const int wave = tid >> 5;
  const int rowBase = (blockIdx.x * WAVES_PER_BLOCK + wave) * ROWS_PER_WAVE;

  const int col = lane & 15;   // A: row; B/C: column
  const int g   = lane >> 4;   // half-wave group

  // ---- A fragments: two 16-row sets, hi & lo planes, 2 chunks of K=32 each.
  v16bf a_hi[2][2], a_lo[2][2];
  #pragma unroll
  for (int s = 0; s < 2; ++s) {
    const float* xr = x + (size_t)(rowBase + s * 16 + col) * D_DIM;
    #pragma unroll
    for (int c = 0; c < 2; ++c) {
      union { v8u u; v16bf b; } H, L;
      #pragma unroll
      for (int grp = 0; grp < 2; ++grp) {
        const float* p = xr + c * 32 + 16 * g + 8 * grp;
        #pragma unroll
        for (int q = 0; q < 4; ++q) {
          float f0 = p[2 * q], f1 = p[2 * q + 1];
          unsigned h0 = f32_to_bf16_rne(f0), h1 = f32_to_bf16_rne(f1);
          unsigned l0 = f32_to_bf16_rne(f0 - bf16_bits_to_f32(h0));
          unsigned l1 = f32_to_bf16_rne(f1 - bf16_bits_to_f32(h1));
          H.u[grp * 4 + q] = h0 | (h1 << 16);
          L.u[grp * 4 + q] = l0 | (l1 << 16);
        }
      }
      a_hi[s][c] = H.b;
      a_lo[s][c] = L.b;
    }
  }

  float    bestVal[2][8];
  unsigned bestIdx[2][8];
  #pragma unroll
  for (int s = 0; s < 2; ++s)
    #pragma unroll
    for (int j = 0; j < 8; ++j) { bestVal[s][j] = 3.0e38f; bestIdx[s][j] = 0u; }

  // ---- LDS chunk pipeline: prefetch chunk 0, then steady state.
  issue_chunk_copy(ecombB, e2, &sbuf[0][0], wave, lane);

  for (int c = 0; c < numChunks; ++c) {
    const int cur = c & 1;
    wait_asynccnt0();            // my copies for chunk c are in LDS
    __syncthreads();             // all copies visible; prev chunk reads done
    if (c + 1 < numChunks)       // prefetch next chunk into other buffer
      issue_chunk_copy(ecombB + (size_t)(c + 1) * CHUNK_BYTES,
                       e2 + (size_t)(c + 1) * CODES_PER_CHUNK,
                       &sbuf[cur ^ 1][0], wave, lane);

    const unsigned char* lb  = &sbuf[cur][0];
    const float*         e2s = (const float*)(lb + CHUNK_BYTES);
    const int chunkBase = c * CODES_PER_CHUNK;

    #pragma unroll
    for (int tt = 0; tt < TILES_PER_CHUNK; ++tt) {
      const unsigned char* rec = lb + (size_t)((tt * 16 + col) * 256 + 32 * g);
      union { v8u u; v16bf b; } h0, h1, l0, l1;
      h0.u = *(const v8u*)(rec + 0);
      h1.u = *(const v8u*)(rec + 64);
      l0.u = *(const v8u*)(rec + 128);
      l1.u = *(const v8u*)(rec + 192);
      const float e2c  = e2s[tt * 16 + col];
      const int   code = chunkBase + tt * 16 + col;

      v8f acc0 = {}, acc1 = {};      // two independent chains interleaved
      acc0 = __builtin_amdgcn_wmma_f32_16x16x32_bf16(false, a_hi[0][0], false, l0.b, (short)0, acc0, false, false);
      acc1 = __builtin_amdgcn_wmma_f32_16x16x32_bf16(false, a_hi[1][0], false, l0.b, (short)0, acc1, false, false);
      acc0 = __builtin_amdgcn_wmma_f32_16x16x32_bf16(false, a_lo[0][0], false, h0.b, (short)0, acc0, false, false);
      acc1 = __builtin_amdgcn_wmma_f32_16x16x32_bf16(false, a_lo[1][0], false, h0.b, (short)0, acc1, false, false);
      acc0 = __builtin_amdgcn_wmma_f32_16x16x32_bf16(false, a_hi[0][0], false, h0.b, (short)0, acc0, false, false);
      acc1 = __builtin_amdgcn_wmma_f32_16x16x32_bf16(false, a_hi[1][0], false, h0.b, (short)0, acc1, false, false);
      acc0 = __builtin_amdgcn_wmma_f32_16x16x32_bf16(false, a_hi[0][1], false, l1.b, (short)0, acc0, false, false);
      acc1 = __builtin_amdgcn_wmma_f32_16x16x32_bf16(false, a_hi[1][1], false, l1.b, (short)0, acc1, false, false);
      acc0 = __builtin_amdgcn_wmma_f32_16x16x32_bf16(false, a_lo[0][1], false, h1.b, (short)0, acc0, false, false);
      acc1 = __builtin_amdgcn_wmma_f32_16x16x32_bf16(false, a_lo[1][1], false, h1.b, (short)0, acc1, false, false);
      acc0 = __builtin_amdgcn_wmma_f32_16x16x32_bf16(false, a_hi[0][1], false, h1.b, (short)0, acc0, false, false);
      acc1 = __builtin_amdgcn_wmma_f32_16x16x32_bf16(false, a_hi[1][1], false, h1.b, (short)0, acc1, false, false);

      #pragma unroll
      for (int j = 0; j < 8; ++j) {
        float s0 = fmaf(-2.0f, acc0[j], e2c);
        if (s0 < bestVal[0][j]) { bestVal[0][j] = s0; bestIdx[0][j] = (unsigned)code; }
        float s1 = fmaf(-2.0f, acc1[j], e2c);
        if (s1 < bestVal[1][j]) { bestVal[1][j] = s1; bestIdx[1][j] = (unsigned)code; }
      }
    }
  }

  // ---- per-row argmin across 16-lane group (tie-break: smaller index),
  // then cooperative fp32 gather of winning code row: 16 lanes x float4.
  #pragma unroll
  for (int s = 0; s < 2; ++s) {
    #pragma unroll
    for (int j = 0; j < 8; ++j) {
      float    bv = bestVal[s][j];
      unsigned bi = bestIdx[s][j];
      #pragma unroll
      for (int m = 1; m < 16; m <<= 1) {
        float    ov = __shfl_xor(bv, m, 32);
        unsigned oi = __shfl_xor(bi, m, 32);
        if (ov < bv || (ov == bv && oi < bi)) { bv = ov; bi = oi; }
      }
      const int row = rowBase + s * 16 + j + 8 * g;
      const v4f* src = (const v4f*)(embed + (size_t)bi * D_DIM) + col;
      v4f*       dst = (v4f*)(out + (size_t)row * D_DIM) + col;
      __builtin_nontemporal_store(*src, dst);
    }
  }
}

// ----------------------------------------------------------------------------
extern "C" void kernel_launch(void* const* d_in, const int* in_sizes, int n_in,
                              void* d_out, int out_size, void* d_ws, size_t ws_size,
                              hipStream_t stream) {
  const float* x     = (const float*)d_in[0];
  const float* embed = (const float*)d_in[1];
  const int nRows = in_sizes[0] / D_DIM;   // 131072
  const int K     = in_sizes[1] / D_DIM;   // 4096

  // ws layout: ecomb[K*128] u16 (hi|lo interleaved) | e2[K] f32  (~1.02 MB)
  unsigned short* ecomb = (unsigned short*)d_ws;
  float*          e2    = (float*)(ecomb + (size_t)K * 128);
  float*          out   = (float*)d_out;

  prep_codebook<<<K, 64, 0, stream>>>(embed, ecomb, e2);

  const int blocks = nRows / (WAVES_PER_BLOCK * ROWS_PER_WAVE);  // 1024
  vq_assign<<<blocks, WAVES_PER_BLOCK * 32, 0, stream>>>(
      x, embed, (const unsigned char*)ecomb, e2, out, K / CODES_PER_CHUNK);
}
